// MultiHeadAttention_4157528342589
// MI455X (gfx1250) — matrix operand order repaired and checked by reading, not compiled, not run on hardware
//
#include <hip/hip_runtime.h>
#include <hip/hip_bf16.h>

typedef __attribute__((ext_vector_type(16))) _Float16 v16h;
typedef __attribute__((ext_vector_type(8)))  _Float16 v8h;
typedef __attribute__((ext_vector_type(8)))  float    v8f;

#define B_ 2
#define S_ 2048
#define H_ 16
#define D_ 64
#define E_ 1024
#define MTOK 4096   // B_*S_

// ---------------------------------------------------------------------------
// WMMA helpers (gfx1250, wave32). Layouts per CDNA5 ISA 7.12.2.
// ---------------------------------------------------------------------------
__device__ __forceinline__ v8f wmma16(v16h a, v16h b, v8f c) {
    // D = A(16x32 f16) * B(32x16 f16) + C(16x16 f32)
    return __builtin_amdgcn_wmma_f32_16x16x32_f16(
        false, a, false, b, (short)0, c, false, false);
}

// A-matrix 16x32 f16 fragment from row-major A (lda elements).
// lane<16: M=lane, K {kb..kb+7, kb+16..kb+23} with kb=0; lane>=16: kb=8.
__device__ __forceinline__ v16h load_a_frag(const _Float16* A, int lda,
                                            int row0, int k0, int lane) {
    const _Float16* p = A + (size_t)(row0 + (lane & 15)) * lda
                          + k0 + ((lane >> 4) << 3);
    v16h f;
    ((v8h*)&f)[0] = *(const v8h*)p;        // K kb .. kb+7
    ((v8h*)&f)[1] = *(const v8h*)(p + 16); // K kb+16 .. kb+23
    return f;
}

// B-matrix 32x16 f16 fragment from BT stored row-major [N, K] (ldb elements).
// lane<16: N=lane, K {0..7, 16..23}; lane>=16: N=lane-16, K {8..15, 24..31}.
__device__ __forceinline__ v16h load_bt_frag(const _Float16* BT, int ldb,
                                             int n0, int k0, int lane) {
    const _Float16* p = BT + (size_t)(n0 + (lane & 15)) * ldb
                           + k0 + ((lane >> 4) << 3);
    v16h f;
    ((v8h*)&f)[0] = *(const v8h*)p;
    ((v8h*)&f)[1] = *(const v8h*)(p + 16);
    return f;
}

// ---------------------------------------------------------------------------
// LayerNorm (fp32 in) -> f16 xn, one token per 256-thread block
// ---------------------------------------------------------------------------
__global__ __launch_bounds__(256)
void ln_kernel(const float* __restrict__ x, const float* __restrict__ g,
               const float* __restrict__ bta, unsigned short* __restrict__ xn_) {
    _Float16* xn = (_Float16*)xn_;
    const int t   = blockIdx.x;
    const int tid = threadIdx.x;
    const float4 v = ((const float4*)(x + (size_t)t * E_))[tid];

    float s = v.x + v.y + v.z + v.w;
#pragma unroll
    for (int o = 16; o >= 1; o >>= 1) s += __shfl_xor(s, o, 32);
    __shared__ float sm[8], sv[8];
    const int wv = tid >> 5, ln = tid & 31;
    if (ln == 0) sm[wv] = s;
    __syncthreads();
    float mean = 0.f;
#pragma unroll
    for (int i = 0; i < 8; ++i) mean += sm[i];
    mean *= (1.0f / E_);

    const float dx = v.x - mean, dy = v.y - mean, dz = v.z - mean, dw = v.w - mean;
    float q = dx * dx + dy * dy + dz * dz + dw * dw;
#pragma unroll
    for (int o = 16; o >= 1; o >>= 1) q += __shfl_xor(q, o, 32);
    if (ln == 0) sv[wv] = q;
    __syncthreads();
    float var = 0.f;
#pragma unroll
    for (int i = 0; i < 8; ++i) var += sv[i];
    var *= (1.0f / E_);
    const float rstd = rsqrtf(var + 1e-5f);

    const int c = tid * 4;
    _Float16* o = xn + (size_t)t * E_ + c;
    o[0] = (_Float16)(dx * rstd * g[c + 0] + bta[c + 0]);
    o[1] = (_Float16)(dy * rstd * g[c + 1] + bta[c + 1]);
    o[2] = (_Float16)(dz * rstd * g[c + 2] + bta[c + 2]);
    o[3] = (_Float16)(dw * rstd * g[c + 3] + bta[c + 3]);
}

// ---------------------------------------------------------------------------
// Transpose + fp32->f16 convert for 1024x1024 weights: WT[n,k] = W[k,n]
// ---------------------------------------------------------------------------
__global__ __launch_bounds__(256)
void tcvt_kernel(const float* __restrict__ W, unsigned short* __restrict__ WT_) {
    _Float16* WT = (_Float16*)WT_;
    const int id = blockIdx.x * 256 + threadIdx.x; // coalesced read
    const int k = id >> 10, n = id & 1023;
    WT[(size_t)n * 1024 + k] = (_Float16)W[id];
}

// ---------------------------------------------------------------------------
// WMMA GEMM: C[M,N] = A[M,K] * BT[N,K]^T (+bias, epilogue per MODE)
//   MODE 0: Q  -> f16 [B,H,S,D], scaled by 1/sqrt(D)=0.125
//   MODE 1: K  -> f16 [B,H,S,D]
//   MODE 2: V  -> f16 [B,H,D,S]  (transposed for P@V B-fragments)
//   MODE 3: out-> f32 [M,N] + bias + residual
// Block: 128 threads = 4 waves (2x2), wave tile 32x64, block tile 64x128.
// ---------------------------------------------------------------------------
template <int MODE>
__global__ __launch_bounds__(128)
void gemm_kernel(const unsigned short* __restrict__ A_,
                 const unsigned short* __restrict__ BT_,
                 const float* __restrict__ bias,
                 const float* __restrict__ resid,
                 void* __restrict__ outv, int Ncols, int Kdim) {
    const _Float16* A  = (const _Float16*)A_;
    const _Float16* BT = (const _Float16*)BT_;
    const int lane = threadIdx.x & 31;
    const int wave = threadIdx.x >> 5;
    const int row0 = blockIdx.y * 64 + (wave >> 1) * 32;
    const int n0   = blockIdx.x * 128 + (wave & 1) * 64;

    v8f acc[2][4] = {};
    for (int k0 = 0; k0 < Kdim; k0 += 32) {
        if (k0 + 32 < Kdim) {
            __builtin_prefetch(A + (size_t)(row0 + (lane & 15)) * Kdim + k0 + 32, 0, 3);
            __builtin_prefetch(BT + (size_t)(n0 + (lane & 15)) * Kdim + k0 + 32, 0, 3);
        }
        const v16h a0 = load_a_frag(A, Kdim, row0,      k0, lane);
        const v16h a1 = load_a_frag(A, Kdim, row0 + 16, k0, lane);
#pragma unroll
        for (int j = 0; j < 4; ++j) {
            const v16h b = load_bt_frag(BT, Kdim, n0 + 16 * j, k0, lane);
            acc[0][j] = wmma16(a0, b, acc[0][j]);
            acc[1][j] = wmma16(a1, b, acc[1][j]);
        }
    }

    const int hi = lane >> 4, nc = lane & 15;
#pragma unroll
    for (int i = 0; i < 2; ++i) {
#pragma unroll
        for (int j = 0; j < 4; ++j) {
            const int n = n0 + 16 * j + nc;
            const float bv = bias[n];
#pragma unroll
            for (int r = 0; r < 8; ++r) {
                const int m = row0 + 16 * i + r + 8 * hi;
                const float val = acc[i][j][r] + bv;
                if constexpr (MODE == 3) {
                    ((float*)outv)[(size_t)m * Ncols + n] =
                        val + resid[(size_t)m * Ncols + n];
                } else {
                    const int bb = m >> 11, s = m & (S_ - 1);   // token -> (b, s)
                    const int hh = n >> 6,  d = n & (D_ - 1);   // col   -> (h, d)
                    _Float16* o = (_Float16*)outv;
                    if constexpr (MODE == 0)
                        o[(((size_t)(bb * H_ + hh)) * S_ + s) * D_ + d] =
                            (_Float16)(val * 0.125f);
                    else if constexpr (MODE == 1)
                        o[(((size_t)(bb * H_ + hh)) * S_ + s) * D_ + d] = (_Float16)val;
                    else
                        o[(((size_t)(bb * H_ + hh)) * D_ + d) * S_ + s] = (_Float16)val;
                }
            }
        }
    }
}

// ---------------------------------------------------------------------------
// Attention: one wave per 16-query tile, online (flash) softmax.
// Mask: key j masked when j < query i (reference masks strict lower triangle),
// so the key loop starts at the query tile and runs to S.
// ---------------------------------------------------------------------------
__global__ __launch_bounds__(128)
void attn_kernel(const unsigned short* __restrict__ Q_,
                 const unsigned short* __restrict__ K_,
                 const unsigned short* __restrict__ Vt_,
                 unsigned short* __restrict__ Hb_) {
    const _Float16* Qh = (const _Float16*)Q_;
    const _Float16* Kh = (const _Float16*)K_;
    const _Float16* Vh = (const _Float16*)Vt_;
    _Float16* Hb = (_Float16*)Hb_;

    __shared__ __align__(32) unsigned short pbuf[4][16 * 32]; // per-wave P tile

    const int lane = threadIdx.x & 31;
    const int wv   = threadIdx.x >> 5;
    const int w    = blockIdx.x * 4 + wv;       // global wave = (bh, qtile)
    const int bh   = w >> 7;                    // / (S_/16)
    const int q0   = (w & 127) * 16;

    const _Float16* Qp = Qh + (size_t)bh * S_ * D_;
    const _Float16* Kp = Kh + (size_t)bh * S_ * D_;
    const _Float16* Vp = Vh + (size_t)bh * D_ * S_;

    const v16h qf0 = load_a_frag(Qp, D_, q0,  0, lane); // d 0..31  (pre-scaled)
    const v16h qf1 = load_a_frag(Qp, D_, q0, 32, lane); // d 32..63

    const int hi = lane >> 4, nc = lane & 15;
    const float NEG = -__builtin_inff();
    float mrow[8], lrow[8];
    v8f acc[4] = {};
#pragma unroll
    for (int r = 0; r < 8; ++r) { mrow[r] = NEG; lrow[r] = 0.f; }

    for (int k0 = (q0 & ~31); k0 < S_; k0 += 32) {
        // ---- scores: two 16x16 tiles over the 32-key chunk (contract D=64) ----
        v8f s0 = {}, s1 = {};
        s0 = wmma16(qf0, load_bt_frag(Kp, D_, k0,       0, lane), s0);
        s0 = wmma16(qf1, load_bt_frag(Kp, D_, k0,      32, lane), s0);
        s1 = wmma16(qf0, load_bt_frag(Kp, D_, k0 + 16,  0, lane), s1);
        s1 = wmma16(qf1, load_bt_frag(Kp, D_, k0 + 16, 32, lane), s1);

        // ---- mask + online softmax (rows live in 16-lane half-groups) ----
#pragma unroll
        for (int r = 0; r < 8; ++r) {
            const int qrow = q0 + r + 8 * hi;
            if (k0 + nc < qrow)      s0[r] = NEG;
            if (k0 + 16 + nc < qrow) s1[r] = NEG;

            float v = fmaxf(s0[r], s1[r]);
#pragma unroll
            for (int o = 1; o < 16; o <<= 1) v = fmaxf(v, __shfl_xor(v, o, 32));
            const float mn   = fmaxf(mrow[r], v);
            const float corr = __expf(mrow[r] - mn);
            mrow[r] = mn;

            const float p0 = __expf(s0[r] - mn);
            const float p1 = __expf(s1[r] - mn);
            s0[r] = p0; s1[r] = p1;
            float rs = p0 + p1;
#pragma unroll
            for (int o = 1; o < 16; o <<= 1) rs += __shfl_xor(rs, o, 32);
            lrow[r] = lrow[r] * corr + rs;
#pragma unroll
            for (int j = 0; j < 4; ++j) acc[j][r] *= corr;
        }

        // ---- C-layout P -> A-layout f16 fragment via per-wave LDS bounce ----
        _Float16* pw = (_Float16*)pbuf[wv];
#pragma unroll
        for (int r = 0; r < 8; ++r) {
            const int M = r + 8 * hi;
            pw[M * 32 + nc]      = (_Float16)s0[r];
            pw[M * 32 + 16 + nc] = (_Float16)s1[r];
        }
        // LDS is in-order per wave; ensure all ds stores land before ds loads.
        asm volatile("s_wait_dscnt 0" ::: "memory");
        const _Float16* pb = pw + (lane & 15) * 32 + ((lane >> 4) << 3);
        v16h pf;
        ((v8h*)&pf)[0] = *(const v8h*)pb;
        ((v8h*)&pf)[1] = *(const v8h*)(pb + 16);

        // ---- acc += P(16x32) @ V(32x64), V stored transposed [D, S] ----
#pragma unroll
        for (int j = 0; j < 4; ++j) {
            const v16h vf = load_bt_frag(Vp, S_, j * 16, k0, lane);
            acc[j] = wmma16(pf, vf, acc[j]);
        }
    }

    // ---- normalize by row sum, write f16 H [token, H*D] ----
    const int bb = bh >> 4, hh = bh & 15;
#pragma unroll
    for (int r = 0; r < 8; ++r) {
        const float inv = 1.0f / lrow[r];
        const int tok = bb * S_ + q0 + r + 8 * hi;
#pragma unroll
        for (int j = 0; j < 4; ++j)
            Hb[(size_t)tok * (H_ * D_) + hh * 64 + j * 16 + nc] =
                (_Float16)(acc[j][r] * inv);
    }
}

// ---------------------------------------------------------------------------
// Launch
// ---------------------------------------------------------------------------
extern "C" void kernel_launch(void* const* d_in, const int* in_sizes, int n_in,
                              void* d_out, int out_size, void* d_ws, size_t ws_size,
                              hipStream_t stream) {
    (void)in_sizes; (void)n_in; (void)out_size; (void)ws_size;
    const float* x    = (const float*)d_in[0];
    const float* ln_g = (const float*)d_in[1];
    const float* ln_b = (const float*)d_in[2];
    const float* wq   = (const float*)d_in[3];
    const float* bq   = (const float*)d_in[4];
    const float* wk   = (const float*)d_in[5];
    const float* bk   = (const float*)d_in[6];
    const float* wv   = (const float*)d_in[7];
    const float* bv   = (const float*)d_in[8];
    const float* w0   = (const float*)d_in[9];
    const float* b0   = (const float*)d_in[10];

    char* ws = (char*)d_ws;
    const size_t MB = 1u << 20;
    unsigned short* xn  = (unsigned short*)(ws);             //  8 MB [4096,1024] f16
    unsigned short* wqT = (unsigned short*)(ws +  8 * MB);   //  2 MB
    unsigned short* wkT = (unsigned short*)(ws + 10 * MB);   //  2 MB
    unsigned short* wvT = (unsigned short*)(ws + 12 * MB);   //  2 MB
    unsigned short* w0T = (unsigned short*)(ws + 14 * MB);   //  2 MB
    unsigned short* Qb  = (unsigned short*)(ws + 16 * MB);   //  8 MB [B,H,S,D]
    unsigned short* Kb  = (unsigned short*)(ws + 24 * MB);   //  8 MB [B,H,S,D]
    unsigned short* Vtb = (unsigned short*)(ws + 32 * MB);   //  8 MB [B,H,D,S]
    unsigned short* Hb  = (unsigned short*)(ws + 40 * MB);   //  8 MB [4096,1024]

    ln_kernel<<<MTOK, 256, 0, stream>>>(x, ln_g, ln_b, xn);
    tcvt_kernel<<<4096, 256, 0, stream>>>(wq, wqT);
    tcvt_kernel<<<4096, 256, 0, stream>>>(wk, wkT);
    tcvt_kernel<<<4096, 256, 0, stream>>>(wv, wvT);
    tcvt_kernel<<<4096, 256, 0, stream>>>(w0, w0T);

    dim3 gg(E_ / 128, MTOK / 64); // (8, 64)
    gemm_kernel<0><<<gg, 128, 0, stream>>>(xn, wqT, bq, nullptr, Qb,  E_, E_);
    gemm_kernel<1><<<gg, 128, 0, stream>>>(xn, wkT, bk, nullptr, Kb,  E_, E_);
    gemm_kernel<2><<<gg, 128, 0, stream>>>(xn, wvT, bv, nullptr, Vtb, E_, E_);

    attn_kernel<<<(B_ * H_ * (S_ / 16)) / 4, 128, 0, stream>>>(Qb, Kb, Vtb, Hb);

    gemm_kernel<3><<<gg, 128, 0, stream>>>(Hb, w0T, b0, x, d_out, E_, E_);
}
